// group_conv_47768626266753
// MI455X (gfx1250) — 8-bit matrix operand order repaired and checked by reading, compiled, not run on hardware
//
#include <hip/hip_runtime.h>


typedef int v8i __attribute__((ext_vector_type(8)));
typedef int vs4i __attribute__((vector_size(16)));
typedef __attribute__((address_space(1))) vs4i* gptr4;
typedef __attribute__((address_space(3))) vs4i* lptr4;

#define BATCH   8
#define CH      256
#define OCH     256
#define HH      64
#define WW      64
#define NBASES  5
#define KTAPS   9
#define WELEM   (OCH*CH*KTAPS)

#if defined(__gfx1250__) && __has_builtin(__builtin_amdgcn_global_load_async_to_lds_b128)
#define HAVE_ASYNC_LDS 1
#else
#define HAVE_ASYNC_LDS 0
#endif

#define XQ_OFF    0
#define WQ_OFF    8388608
#define PART_OFF  11337728
#define ALPHA_OFF 11340608

__global__ void wabs_partial(const float* __restrict__ w, float* __restrict__ part) {
    __shared__ float sm[256];
    int b   = blockIdx.x / 144;
    int seg = blockIdx.x % 144;
    const float* p = w + (size_t)b * WELEM + (size_t)seg * 4096;
    float s = 0.f;
    #pragma unroll
    for (int i = 0; i < 16; ++i) s += fabsf(p[i * 256 + threadIdx.x]);
    sm[threadIdx.x] = s;
    __syncthreads();
    for (int off = 128; off > 0; off >>= 1) {
        if ((int)threadIdx.x < off) sm[threadIdx.x] += sm[threadIdx.x + off];
        __syncthreads();
    }
    if (threadIdx.x == 0) part[blockIdx.x] = sm[0];
}

__global__ void alpha_finish(const float* __restrict__ part, float* __restrict__ alpha) {
    __shared__ float sm[256];
    int b = blockIdx.x;
    float s = ((int)threadIdx.x < 144) ? part[b * 144 + threadIdx.x] : 0.f;
    sm[threadIdx.x] = s;
    __syncthreads();
    for (int off = 128; off > 0; off >>= 1) {
        if ((int)threadIdx.x < off) sm[threadIdx.x] += sm[threadIdx.x + off];
        __syncthreads();
    }
    if (threadIdx.x == 0) alpha[b] = sm[0] * (1.0f / (float)WELEM);
}

__global__ void quant_x(const float* __restrict__ x, unsigned char* __restrict__ xq) {
    __shared__ alignas(16) unsigned char sm[16][256];
    int blk = blockIdx.x;
    int w16 = blk & 3;
    int h   = (blk >> 2) & 63;
    int n   = blk >> 8;
    int t   = threadIdx.x;
    int wl  = t & 15;
    int cg  = t >> 4;
    int w0  = w16 * 16;
    const float* xp = x + (((size_t)n * CH) * HH + h) * WW + w0 + wl;
    #pragma unroll
    for (int i = 0; i < 16; ++i) {
        int c = cg + i * 16;
        float v = xp[(size_t)c * HH * WW];
        sm[wl][c] = (v > 0.5f) ? (unsigned char)1 : (unsigned char)0;
    }
    __syncthreads();
    int wl2 = t >> 4;
    int c16 = (t & 15) * 16;
    uint4 d = *(const uint4*)&sm[wl2][c16];
    *(uint4*)(xq + ((((size_t)n * HH + h) * WW) + w0 + wl2) * CH + c16) = d;
}

__global__ void quant_w(const float* __restrict__ w, unsigned char* __restrict__ wq) {
    int idx = blockIdx.x * 256 + threadIdx.x;
    int c4  = idx & 63;
    int o   = (idx >> 6) & 255;
    int tap = (idx >> 14) % 9;
    int b   = idx / (64 * 256 * 9);
    const float* p = w + (((size_t)(b * 256 + o) * 256) + c4 * 4) * 9 + tap;
    unsigned int u = 0;
    #pragma unroll
    for (int j = 0; j < 4; ++j) {
        float v = p[j * 9];
        unsigned int s = (v >= 0.f) ? 0x01u : 0xFFu;
        u |= s << (8 * j);
    }
    ((unsigned int*)wq)[idx] = u;
}

__global__ __launch_bounds__(128) void bconv_wmma(
        const unsigned char* __restrict__ xq,
        const unsigned char* __restrict__ wq,
        const float* __restrict__ alpha,
        float* __restrict__ out)
{
    __shared__ alignas(16) unsigned char bsm[NBASES * 16 * 256];
    __shared__ float tile[4][16][16];

    const int lane = threadIdx.x & 31;
    const int wv   = threadIdx.x >> 5;
    const int o0     = (blockIdx.x & 15) * 16;
    const int mgroup = blockIdx.x >> 4;
    const int ms     = mgroup * 4 + wv;
    const int n    = ms >> 7;
    const int oh   = (ms >> 1) & 63;
    const int ow0  = (ms & 1) * 32;

    const int half = lane >> 4;
    const int l15  = lane & 15;

    v8i acc[2][NBASES];
    #pragma unroll
    for (int t = 0; t < 2; ++t)
        #pragma unroll
        for (int b = 0; b < NBASES; ++b)
            #pragma unroll
            for (int r = 0; r < 8; ++r) acc[t][b][r] = 0;

    for (int tap = 0; tap < 9; ++tap) {
        const int kh = tap / 3, kw = tap - kh * 3;
        const int ih = oh + kh - 1;
        const bool rowok = ((unsigned)ih < 64u);

        const unsigned char* arow[2];
        bool aok[2];
        #pragma unroll
        for (int t = 0; t < 2; ++t) {
            int iw = ow0 + t * 16 + l15 + kw - 1;
            aok[t] = rowok && ((unsigned)iw < 64u);
            int iwc = iw < 0 ? 0 : (iw > 63 ? 63 : iw);
            int ihc = ih < 0 ? 0 : (ih > 63 ? 63 : ih);
            arow[t] = xq + ((((size_t)n * HH + ihc) * WW) + iwc) * CH;
            if (aok[t]) __builtin_prefetch(arow[t], 0, 0);
        }

        #pragma unroll
        for (int i = 0; i < 10; ++i) {
            int f   = i * 2048 + (int)threadIdx.x * 16;
            int b   = f >> 12;
            int rem = f & 4095;
            int ol  = rem >> 8;
            int c   = rem & 255;
            const unsigned char* g =
                wq + (size_t)(b * 9 + tap) * 65536 + (size_t)(o0 + ol) * 256 + c;
#if HAVE_ASYNC_LDS
            __builtin_amdgcn_global_load_async_to_lds_b128(
                (gptr4)(void*)g, (lptr4)(void*)&bsm[f], 0, 0);
#else
            *(uint4*)&bsm[f] = *(const uint4*)g;
#endif
        }
#if HAVE_ASYNC_LDS
        asm volatile("s_wait_asynccnt 0x0" ::: "memory");
#endif
        __syncthreads();

        #pragma unroll
        for (int ck = 0; ck < 4; ++ck) {
            const int c0 = ck * 64;

            v8i afrag[2];
            #pragma unroll
            for (int t = 0; t < 2; ++t) {
                #pragma unroll
                for (int v = 0; v < 4; ++v) {
                    uint2 d = make_uint2(0u, 0u);
                    if (aok[t]) d = *(const uint2*)(arow[t] + c0 + 32 * (v >> 1) + 8 * (v & 1) + 16 * half);
                    afrag[t][2 * v]     = (int)d.x;
                    afrag[t][2 * v + 1] = (int)d.y;
                }
            }

            #pragma unroll
            for (int b = 0; b < NBASES; ++b) {
                const unsigned char* bp = &bsm[(b * 16 + l15) * 256 + c0 + 16 * half];
                uint4 q0 = *(const uint4*)(bp);
                uint4 q1 = *(const uint4*)(bp + 32);
                v8i bfrag;
                bfrag[0] = (int)q0.x; bfrag[1] = (int)q0.y;
                bfrag[2] = (int)q0.z; bfrag[3] = (int)q0.w;
                bfrag[4] = (int)q1.x; bfrag[5] = (int)q1.y;
                bfrag[6] = (int)q1.z; bfrag[7] = (int)q1.w;

                #pragma unroll
                for (int t = 0; t < 2; ++t) {
                    acc[t][b] = __builtin_amdgcn_wmma_i32_16x16x64_iu8(
                        false, afrag[t], true, bfrag, acc[t][b], false, false);
                }
            }
        }
        __syncthreads();
    }

    const float a0 = alpha[0] * 0.2f, a1 = alpha[1] * 0.2f, a2 = alpha[2] * 0.2f,
                a3 = alpha[3] * 0.2f, a4 = alpha[4] * 0.2f;

    #pragma unroll
    for (int t = 0; t < 2; ++t) {
        #pragma unroll
        for (int r = 0; r < 8; ++r) {
            float s = a0 * (float)acc[t][0][r] + a1 * (float)acc[t][1][r]
                    + a2 * (float)acc[t][2][r] + a3 * (float)acc[t][3][r]
                    + a4 * (float)acc[t][4][r];
            tile[wv][l15][r + 8 * half] = s;
        }
        __syncthreads();
        #pragma unroll
        for (int it = 0; it < 8; ++it) {
            int ol = 2 * it + half;
            float v = tile[wv][ol][l15];
            out[(((size_t)n * OCH + o0 + ol) * HH + oh) * WW + ow0 + t * 16 + l15] = v;
        }
        __syncthreads();
    }
}

extern "C" void kernel_launch(void* const* d_in, const int* in_sizes, int n_in,
                              void* d_out, int out_size, void* d_ws, size_t ws_size,
                              hipStream_t stream) {
    const float* x = (const float*)d_in[0];
    const float* w = (const float*)d_in[1];
    unsigned char* ws = (unsigned char*)d_ws;

    unsigned char* xq    = ws + XQ_OFF;
    unsigned char* wqb   = ws + WQ_OFF;
    float*         part  = (float*)(ws + PART_OFF);
    float*         alpha = (float*)(ws + ALPHA_OFF);
    float*         out   = (float*)d_out;

    wabs_partial<<<NBASES * 144, 256, 0, stream>>>(w, part);
    alpha_finish<<<NBASES, 256, 0, stream>>>(part, alpha);
    quant_x<<<BATCH * HH * (WW / 16), 256, 0, stream>>>(x, xq);
    quant_w<<<2880, 256, 0, stream>>>(w, wqb);
    bconv_wmma<<<4096, 128, 0, stream>>>(xq, wqb, alpha, out);
}
